// AutoRegressiveLSTMEncoder_71674414235850
// MI455X (gfx1250) — matrix operand order repaired and checked by reading, not compiled, not run on hardware
//
#include <hip/hip_runtime.h>

// ---------------------------------------------------------------------------
// MI455X (gfx1250) autoregressive LSTM encoder.
//   ~26.7 TFLOP of bf16 WMMA; weights (24 MB bf16) are L2-resident.
//   M=32 batch rows per workgroup doubles B-fragment reuse (each weight
//   fragment feeds two M-tiles), halving L2 weight traffic vs M=16 and
//   balancing L2 bandwidth against the v_wmma_f32_16x16x32_bf16 pipes.
// Workgroup = 32 batch rows x 512 threads (16 waves, wave32); wave owns 32
// gate columns (2 N-tiles) x 2 M-tiles. t_h / e / h live in LDS (~276 KB,
// padded rows for conflict-free ds_load_b128); cell state c in VGPRs.
// Weight bases are offset by an opaque zero each step so LICM cannot hoist
// the (d-invariant) B-fragment loads out of the 32-step loop and spill them.
// ---------------------------------------------------------------------------

typedef __attribute__((ext_vector_type(8)))  __bf16 v8bf;
typedef __attribute__((ext_vector_type(16))) __bf16 v16bf;
typedef __attribute__((ext_vector_type(8)))  float  v8f;

#define NB    32768      // batch
#define NE    300        // emb dim
#define ND    32         // digits (steps)
#define NA    64         // ary
#define NH    1024       // hidden
#define NCAT  2048       // H + EC
#define LDH   1032       // padded LDS row stride (bank step 4)
#define MROW  32         // batch rows per workgroup

// workspace layout (bytes)
#define OFF_WIH   0ull                         // [4096][2048] bf16 = 16 MB
#define OFF_WHH   16777216ull                  // [4096][1024] bf16 =  8 MB
#define OFF_WHZ   25165824ull                  // [64][1024]   bf16
#define OFF_WEMB  25296896ull                  // [1024][64]   bf16
#define OFF_TH    25427968ull                  // [32768][1024] bf16 = 64 MB

#define WMMAB(A, Bf, C) \
  __builtin_amdgcn_wmma_f32_16x16x32_bf16(false, (A), false, (Bf), (short)0, (C), false, false)

static __device__ __forceinline__ float sigm(float x) { return 1.f / (1.f + __expf(-x)); }

// A fragment (16x32 bf16, M x K): lane L -> row m = L&15; hi = L>>4 selects
// K-halves: elements i map to k = k0 + hi*16 + i  (two 16B loads)
static __device__ __forceinline__ v16bf load_a(const __bf16* __restrict__ p) {
  v8bf lo = *(const v8bf*)p;
  v8bf hh = *(const v8bf*)(p + 8);
  v16bf r;
#pragma unroll
  for (int i = 0; i < 8; ++i) { r[i] = lo[i]; r[i + 8] = hh[i]; }
  return r;
}

// B fragment (32x16 bf16, K x N): lane L -> col n = L&15, elements i -> k =
// k0 + (L>>4)*16 + i. B[k][n] = W[n][k] row-major -> one contiguous 32B load.
static __device__ __forceinline__ v16bf load_b(const __bf16* __restrict__ p) {
  return *(const v16bf*)p;
}

// ---------------------------------------------------------------------------
// Kernel 0: fp32 -> bf16 weight conversion into workspace
// ---------------------------------------------------------------------------
#define CN1 (4096 * 2048)
#define CN2 (4096 * 1024)
#define CN3 (64 * 1024)
#define CN4 (1024 * 64)
#define CNT (CN1 + CN2 + CN3 + CN4)

__global__ void k_convert(const float* __restrict__ wih, const float* __restrict__ whh,
                          const float* __restrict__ whz, const float* __restrict__ wemb,
                          __bf16* __restrict__ o_ih, __bf16* __restrict__ o_hh,
                          __bf16* __restrict__ o_hz, __bf16* __restrict__ o_emb) {
  long i = (long)blockIdx.x * blockDim.x + threadIdx.x;
  if (i >= CNT) return;
  if (i < CN1)                  o_ih[i]                    = (__bf16)wih[i];
  else if (i < CN1 + CN2)       o_hh[i - CN1]              = (__bf16)whh[i - CN1];
  else if (i < CN1 + CN2 + CN3) o_hz[i - CN1 - CN2]        = (__bf16)whz[i - CN1 - CN2];
  else                          o_emb[i - CN1 - CN2 - CN3] = (__bf16)wemb[i - CN1 - CN2 - CN3];
}

// ---------------------------------------------------------------------------
// Kernel 1: t_h = tanh(x @ W_xh^T + b_xh), bf16 out.  (~0.07% of total FLOPs)
// ---------------------------------------------------------------------------
__global__ __launch_bounds__(256) void k_input(const float* __restrict__ x,
                                               const float* __restrict__ wxh,
                                               const float* __restrict__ bxh,
                                               __bf16* __restrict__ th) {
  __shared__ float sx[304];
  const int b = blockIdx.x;
  for (int i = threadIdx.x; i < NE; i += 256) sx[i] = x[(long)b * NE + i];
  __syncthreads();
  for (int h = threadIdx.x; h < NH; h += 256) {
    const float* wr = wxh + (long)h * NE;
    float acc = bxh[h];
#pragma unroll 4
    for (int k = 0; k < NE; ++k) acc += sx[k] * wr[k];
    th[(long)b * NH + h] = (__bf16)tanhf(acc);
  }
}

// ---------------------------------------------------------------------------
// Kernel 2: 32 autoregressive LSTM steps for a 32-row batch tile.
// ---------------------------------------------------------------------------
// One K-step: 2 A-fragments (two M-tiles) x 4 B-fragments (gates) -> 8 WMMA.
#define GSTEP(A0P, A1P, WPTR, WLD, WOFF)                                            \
  {                                                                                 \
    v16bf A0 = load_a(A0P);                                                         \
    v16bf A1 = load_a(A1P);                                                         \
    v16bf B0 = load_b((WPTR) + (long)(j0 + nl) * (WLD) + (WOFF) + hi * 16);         \
    v16bf B1 = load_b((WPTR) + (long)(1024 + j0 + nl) * (WLD) + (WOFF) + hi * 16);  \
    v16bf B2 = load_b((WPTR) + (long)(2048 + j0 + nl) * (WLD) + (WOFF) + hi * 16);  \
    v16bf B3 = load_b((WPTR) + (long)(3072 + j0 + nl) * (WLD) + (WOFF) + hi * 16);  \
    ai0 = WMMAB(A0, B0, ai0); ai1 = WMMAB(A1, B0, ai1);                             \
    af0 = WMMAB(A0, B1, af0); af1 = WMMAB(A1, B1, af1);                             \
    ag0 = WMMAB(A0, B2, ag0); ag1 = WMMAB(A1, B2, ag1);                             \
    ao0 = WMMAB(A0, B3, ao0); ao1 = WMMAB(A1, B3, ao1);                             \
  }

__global__ __launch_bounds__(512) void k_lstm(const __bf16* __restrict__ g_th,
                                              const __bf16* __restrict__ Wih,
                                              const __bf16* __restrict__ Whh,
                                              const __bf16* __restrict__ Whz,
                                              const __bf16* __restrict__ Wemb,
                                              const float* __restrict__ b_ih,
                                              const float* __restrict__ b_hh,
                                              const float* __restrict__ b_hz,
                                              float* __restrict__ out,
                                              int oz /* always 0: defeats LICM */) {
  __shared__ __align__(16) __bf16 sm_th[MROW * LDH];   //  66 KB: tanh(x W_xh^T)
  __shared__ __align__(16) __bf16 sm_h[2][MROW * LDH]; // 132 KB: h ping-pong
  __shared__ __align__(16) __bf16 sm_e[MROW * LDH];    //  66 KB: e
  __shared__ __align__(16) float  sm_z[MROW * NA];     //   8 KB: pre-softmax z
  __shared__ __align__(16) __bf16 sm_p[MROW * NA];     //   4 KB: softmax probs

  const int tid  = threadIdx.x;
  const int lane = tid & 31;
  const int wv   = tid >> 5;       // 16 waves
  const int nl   = lane & 15;      // N index (C/D layout) / M row (A layout)
  const int hi   = lane >> 4;      // half-select
  const int b0   = blockIdx.x * MROW;

  for (int i = tid; i < MROW * NH; i += 512) {
    const int row = i >> 10, col = i & (NH - 1);
    sm_th[row * LDH + col]   = g_th[(long)(b0 + row) * NH + col];
    sm_h[0][row * LDH + col] = (__bf16)0.f;
  }
  __syncthreads();

  v8f c_reg[2][2] = {};            // cell state: [n-tile][m-tile] x 8 rows/lane

#pragma unroll 1
  for (int d = 0; d < ND; ++d) {
    // Opaque zero offset (oz == 0) makes these bases loop-VARIANT so LICM
    // cannot hoist the d-invariant fragment loads and spill them to scratch.
    const int dz = d & oz;
    const __bf16* thD   = sm_th + dz;
    const __bf16* WihD  = Wih  + dz;
    const __bf16* WhhD  = Whh  + dz;
    const __bf16* WhzD  = Whz  + dz;
    const __bf16* WembD = Wemb + dz;

    const __bf16* h_cur = sm_h[d & 1];
    __bf16*       h_nxt = sm_h[(d & 1) ^ 1];

    // ---- phase 1: gates = [t_h,e] @ W_ih^T + h @ W_hh^T
    //      wave owns 32 cols x both 16-row M-tiles; B fragments reused 2x.
#pragma unroll
    for (int nt = 0; nt < 2; ++nt) {
      const int j0 = wv * 32 + nt * 16;
      v8f ai0 = {}, af0 = {}, ag0 = {}, ao0 = {};
      v8f ai1 = {}, af1 = {}, ag1 = {}, ao1 = {};
      for (int ks = 0; ks < 32; ++ks) {            // t_h part, K = 0..1023
        const int k0 = ks * 32;
        GSTEP(thD + nl * LDH + k0 + hi * 16,
              thD + (16 + nl) * LDH + k0 + hi * 16, WihD, NCAT, k0);
      }
      if (d > 0) {                                 // e part, K = 1024..2047 (e0==0)
        for (int ks = 0; ks < 32; ++ks) {
          const int k0 = ks * 32;
          GSTEP(sm_e + nl * LDH + k0 + hi * 16,
                sm_e + (16 + nl) * LDH + k0 + hi * 16, WihD, NCAT, 1024 + k0);
        }
      }
      for (int ks = 0; ks < 32; ++ks) {            // recurrent part
        const int k0 = ks * 32;
        GSTEP(h_cur + nl * LDH + k0 + hi * 16,
              h_cur + (16 + nl) * LDH + k0 + hi * 16, WhhD, NH, k0);
      }
      const int jc = j0 + nl;
      const float bi = b_ih[jc]        + b_hh[jc];
      const float bf = b_ih[1024 + jc] + b_hh[1024 + jc];
      const float bg = b_ih[2048 + jc] + b_hh[2048 + jc];
      const float bo = b_ih[3072 + jc] + b_hh[3072 + jc];
#pragma unroll
      for (int r = 0; r < 8; ++r) {                // M-tile 0: row r + hi*8
        const float iv = sigm(ai0[r] + bi);
        const float fv = sigm(af0[r] + bf);
        const float gv = tanhf(ag0[r] + bg);
        const float ov = sigm(ao0[r] + bo);
        const float c  = fv * c_reg[nt][0][r] + iv * gv;
        c_reg[nt][0][r] = c;
        h_nxt[(r + hi * 8) * LDH + jc] = (__bf16)(ov * tanhf(c));
      }
#pragma unroll
      for (int r = 0; r < 8; ++r) {                // M-tile 1: row 16 + r + hi*8
        const float iv = sigm(ai1[r] + bi);
        const float fv = sigm(af1[r] + bf);
        const float gv = tanhf(ag1[r] + bg);
        const float ov = sigm(ao1[r] + bo);
        const float c  = fv * c_reg[nt][1][r] + iv * gv;
        c_reg[nt][1][r] = c;
        h_nxt[(16 + r + hi * 8) * LDH + jc] = (__bf16)(ov * tanhf(c));
      }
    }
    __syncthreads();

    // ---- phase 2: z = log(softplus(h @ W_hz^T + b_hz) + eps)  (waves 0..7)
    if (wv < 8) {
      const int mt = wv >> 2;                      // M-tile
      const int n0 = (wv & 3) * 16;                // z column tile
      v8f az = {};
      for (int ks = 0; ks < 32; ++ks) {
        const int k0 = ks * 32;
        v16bf Afr = load_a(h_nxt + (mt * 16 + nl) * LDH + k0 + hi * 16);
        az = WMMAB(Afr, load_b(WhzD + (long)(n0 + nl) * NH + k0 + hi * 16), az);
      }
      const float bz = b_hz[n0 + nl];
#pragma unroll
      for (int r = 0; r < 8; ++r) {
        const float v  = az[r] + bz;
        const float sp = fmaxf(v, 0.f) + log1pf(__expf(-fabsf(v)));
        sm_z[(mt * 16 + r + hi * 8) * NA + n0 + nl] = __logf(sp + 1e-6f);
      }
    }
    __syncthreads();

    // ---- phase 3: per-row softmax over A=64; emit probs (both tuple copies)
    if (tid < MROW) {
      const int row = tid;
      float mx = -3.4e38f;
      for (int a = 0; a < NA; ++a) mx = fmaxf(mx, sm_z[row * NA + a]);
      float s = 0.f;
      for (int a = 0; a < NA; ++a) s += __expf(sm_z[row * NA + a] - mx);
      const float inv = 1.f / s;
      float* o0 = out + (long)(b0 + row) * (ND * NA) + (long)d * NA;
      float* o1 = o0 + (long)NB * ND * NA;
      for (int a = 0; a < NA; ++a) {
        const float p = __expf(sm_z[row * NA + a] - mx) * inv;
        sm_p[row * NA + a] = (__bf16)p;
        o0[a] = p;
        o1[a] = p;
      }
    }
    __syncthreads();

    // ---- phase 4: e = p @ W_emb^T  (K = 64, 2 WMMA k-steps, both M-tiles)
    for (int t4 = 0; t4 < 4; ++t4) {
      const int n0 = wv * 64 + t4 * 16;
#pragma unroll
      for (int mt = 0; mt < 2; ++mt) {
        v8f ae = {};
#pragma unroll
        for (int ks = 0; ks < 2; ++ks) {
          const int k0 = ks * 32;
          v16bf Afr = load_a(sm_p + (mt * 16 + nl) * NA + k0 + hi * 16);
          ae = WMMAB(Afr, load_b(WembD + (long)(n0 + nl) * NA + k0 + hi * 16), ae);
        }
#pragma unroll
        for (int r = 0; r < 8; ++r)
          sm_e[(mt * 16 + r + hi * 8) * LDH + n0 + nl] = (__bf16)ae[r];
      }
    }
    __syncthreads();                               // e visible before next step
  }
}

// ---------------------------------------------------------------------------
extern "C" void kernel_launch(void* const* d_in, const int* in_sizes, int n_in,
                              void* d_out, int out_size, void* d_ws, size_t ws_size,
                              hipStream_t stream) {
  (void)in_sizes; (void)n_in; (void)out_size; (void)ws_size;
  const float* x    = (const float*)d_in[0];
  const float* Wxh  = (const float*)d_in[1];
  const float* bxh  = (const float*)d_in[2];
  const float* Wih  = (const float*)d_in[3];
  const float* Whh  = (const float*)d_in[4];
  const float* bih  = (const float*)d_in[5];
  const float* bhh  = (const float*)d_in[6];
  const float* Whz  = (const float*)d_in[7];
  const float* bhz  = (const float*)d_in[8];
  const float* Wemb = (const float*)d_in[9];
  float* out = (float*)d_out;

  char* ws = (char*)d_ws;
  __bf16* Wih_bf  = (__bf16*)(ws + OFF_WIH);
  __bf16* Whh_bf  = (__bf16*)(ws + OFF_WHH);
  __bf16* Whz_bf  = (__bf16*)(ws + OFF_WHZ);
  __bf16* Wemb_bf = (__bf16*)(ws + OFF_WEMB);
  __bf16* th_bf   = (__bf16*)(ws + OFF_TH);

  k_convert<<<(CNT + 255) / 256, 256, 0, stream>>>(Wih, Whh, Whz, Wemb,
                                                   Wih_bf, Whh_bf, Whz_bf, Wemb_bf);
  k_input<<<NB, 256, 0, stream>>>(x, Wxh, bxh, th_bf);
  k_lstm<<<NB / MROW, 512, 0, stream>>>(th_bf, Wih_bf, Whh_bf, Whz_bf, Wemb_bf,
                                        bih, bhh, bhz, out, /*oz=*/0);
}
